// SWGATLayer_28235115003924
// MI455X (gfx1250) — matrix operand order repaired and checked by reading, not compiled, not run on hardware
//
#include <hip/hip_runtime.h>

typedef __attribute__((ext_vector_type(16))) __bf16 v16bf;
typedef __attribute__((ext_vector_type(4)))  __bf16 bf4;
typedef __attribute__((ext_vector_type(8)))  float  v8f;
typedef __attribute__((ext_vector_type(4)))  float  f4;
typedef __attribute__((ext_vector_type(2)))  float  f2;

// scalar float -> bf16 RNE (used only in one-time setup paths)
__device__ __forceinline__ __bf16 f2bf(float f) {
  unsigned b = __float_as_uint(f);
  unsigned r = (b + 0x7FFFu + ((b >> 16) & 1u)) >> 16;
  unsigned short h = (unsigned short)r;
  __bf16 o;
  __builtin_memcpy(&o, &h, sizeof(o));
  return o;
}

// monotone float <-> uint mapping so unsigned atomicMax == float max
__device__ __forceinline__ unsigned fkey(float f) {
  unsigned b = __float_as_uint(f);
  return b ^ ((unsigned)((int)b >> 31) | 0x80000000u);
}
__device__ __forceinline__ float funkey(unsigned u) {
  unsigned b = (u & 0x80000000u) ? (u ^ 0x80000000u) : ~u;
  return __uint_as_float(b);
}

// ---------------------------------------------------------------- init
__global__ void init_kernel(float* __restrict__ out, float* __restrict__ denom,
                            unsigned* __restrict__ mkey, int n_dst) {
  int stride = gridDim.x * blockDim.x;
  int total = n_dst * 66;
  for (int i = blockIdx.x * blockDim.x + threadIdx.x; i < total; i += stride) {
    if (i < n_dst * 64)      out[i] = 0.f;
    else if (i < n_dst * 65) denom[i - n_dst * 64] = 0.f;
    else                     mkey[i - n_dst * 65] = 0u;   // below any real key
  }
}

// ------------------------------------------- small dense projections (N=64)
__global__ void __launch_bounds__(256)
proj_kernel(const float* __restrict__ X, const float* __restrict__ W,
            float* __restrict__ Z, int nrows, int K) {
  __shared__ float xs[4 * 300];              // 4 rows, K <= 300
  const int t = threadIdx.x;
  const int rb = t >> 6;                     // row within block
  const int col = t & 63;
  const long row0 = (long)blockIdx.x * 4;
  for (int i = t; i < 4 * K; i += 256) {
    const int rr = i / K;
    const long row = row0 + rr;
    xs[i] = (row < nrows) ? X[row * K + (i - rr * K)] : 0.f;
  }
  __syncthreads();
  const long row = row0 + rb;
  if (row < nrows) {
    float acc = 0.f;
    for (int k = 0; k < K; ++k)
      acc = fmaf(xs[rb * K + k], W[k * 64 + col], acc);
    Z[row * 64 + col] = acc;
  }
}

// ------------------------------- WMMA edge kernel: logits + segment max
// Per wave: 16-edge tile. dfeat(16x64) = tfidf_tile(16x64) x W_feat(64x64)
// via 8x v_wmma_f32_16x16x32_bf16, then fused gather-add, leaky, attn dot.
__global__ void __launch_bounds__(256)
edge_logits_kernel(const float* __restrict__ tfidf,
                   const float* __restrict__ W_feat,   // [64,64] row-major
                   const float* __restrict__ b_feat,   // [64]
                   const float* __restrict__ W_attn,   // [64]
                   const float* __restrict__ z_src,    // [N_SRC,64]
                   const float* __restrict__ z_dst,    // [N_DST,64]
                   const int* __restrict__ edge_src,
                   const int* __restrict__ edge_dst,
                   float* __restrict__ logits,
                   unsigned* __restrict__ mkey,
                   int E) {
  __shared__ float zsum[8 * 16 * 65];        // per-wave 16x64 (stride 65)
  const int lane = threadIdx.x & 31;
  const int wib  = threadIdx.x >> 5;
  const int col  = lane & 15;
  const int half = lane >> 4;
  float* zs = zsum + wib * (16 * 65);

  // Persistent B: W_feat as bf16 32x16 B operands in contiguous k order (matches A), 4 N-tiles x 2 K-steps.
  // lane half h holds K = kstep*32 + h*16 + 2j{,+1}, n = t*16 + col.
  v16bf B[4][2];
#pragma unroll
  for (int t = 0; t < 4; ++t)
#pragma unroll
    for (int s = 0; s < 2; ++s)
#pragma unroll
      for (int j = 0; j < 8; ++j) {
        const int k0 = s * 32 + half * 16 + 2 * j;
        const int n  = t * 16 + col;
        B[t][s][2 * j]     = f2bf(W_feat[(size_t)k0 * 64 + n]);
        B[t][s][2 * j + 1] = f2bf(W_feat[(size_t)(k0 + 1) * 64 + n]);
      }

  float wa[4];
#pragma unroll
  for (int t = 0; t < 4; ++t) wa[t] = W_attn[t * 16 + col];
  const float bf0 = b_feat[2 * lane];
  const float bf1 = b_feat[2 * lane + 1];

  const int ntiles = (E + 15) >> 4;
  const int gwave  = blockIdx.x * (blockDim.x >> 5) + wib;
  const int nwaves = gridDim.x * (blockDim.x >> 5);

  for (int tile = gwave; tile < ntiles; tile += nwaves) {
    const int base = tile << 4;
    const int m    = col;                      // this lane's edge row in tile
    const int eidx = min(base + m, E - 1);
    const int my_src = edge_src[eidx];
    const int my_dst = edge_dst[eidx];

    // zsum[m][d] = z_src[src_m][d] + z_dst[dst_m][d] + b_feat[d]
#pragma unroll 4
    for (int mm = 0; mm < 16; ++mm) {
      const int s_idx = __shfl(my_src, mm);
      const int d_idx = __shfl(my_dst, mm);
      const f2 vs = ((const f2*)(z_src + (size_t)s_idx * 64))[lane];
      const f2 vd = ((const f2*)(z_dst + (size_t)d_idx * 64))[lane];
      zs[mm * 65 + 2 * lane]     = vs.x + vd.x + bf0;
      zs[mm * 65 + 2 * lane + 1] = vs.y + vd.y + bf1;
    }

    // A tile as bf16 16x32 A operands in contiguous k order (same as B): lane half h holds
    // K = kstep*32 + h*16 + q (elems 0..15).
    // Vector fptrunc -> packed v_cvt bf16 conversions (hot path).
    const float* arow = tfidf + (size_t)eidx * 64;
    v16bf A[2];
#pragma unroll
    for (int s = 0; s < 2; ++s) {
      const int kb = s * 32 + half * 16;
      const f4 c0 = __builtin_nontemporal_load((const f4*)(arow + kb));
      const f4 c1 = __builtin_nontemporal_load((const f4*)(arow + kb + 4));
      const f4 c2 = __builtin_nontemporal_load((const f4*)(arow + kb + 8));
      const f4 c3 = __builtin_nontemporal_load((const f4*)(arow + kb + 12));
      union { v16bf v; bf4 q[4]; } au;
      au.q[0] = __builtin_convertvector(c0, bf4);
      au.q[1] = __builtin_convertvector(c1, bf4);
      au.q[2] = __builtin_convertvector(c2, bf4);
      au.q[3] = __builtin_convertvector(c3, bf4);
      A[s] = au.v;
    }

    v8f C[4];
#pragma unroll
    for (int t = 0; t < 4; ++t) C[t] = 0.f;
#pragma unroll
    for (int t = 0; t < 4; ++t) {
      C[t] = __builtin_amdgcn_wmma_f32_16x16x32_bf16(
          false, A[0], false, B[t][0], (short)0, C[t], false, false);
      C[t] = __builtin_amdgcn_wmma_f32_16x16x32_bf16(
          false, A[1], false, B[t][1], (short)0, C[t], false, false);
    }

    // C layout: lane holds D[r + 8*half][col] in element r of each C[t].
    float part[8];
#pragma unroll
    for (int r = 0; r < 8; ++r) part[r] = 0.f;
#pragma unroll
    for (int t = 0; t < 4; ++t)
#pragma unroll
      for (int r = 0; r < 8; ++r) {
        const float z2 = C[t][r] + zs[(r + 8 * half) * 65 + t * 16 + col];
        const float z3 = z2 > 0.f ? z2 : 0.01f * z2;
        part[r] = fmaf(z3, wa[t], part[r]);
      }
    // reduce over the 16 lanes of each half (N dimension)
#pragma unroll
    for (int r = 0; r < 8; ++r) {
      part[r] += __shfl_xor(part[r], 1);
      part[r] += __shfl_xor(part[r], 2);
      part[r] += __shfl_xor(part[r], 4);
      part[r] += __shfl_xor(part[r], 8);
    }
    // writer lanes: half0 cols 0..7 -> edges 0..7; half1 cols 8..15 -> edges 8..15
    const int r_own = col - 8 * half;
    if (r_own >= 0 && r_own < 8) {
      const int edge = base + m;               // m == r_own + 8*half
      if (edge < E) {
        const float lg = part[r_own];
        logits[edge] = lg;
        atomicMax(mkey + my_dst, fkey(lg));
      }
    }
  }
}

// ----------------- pass 2: exp, denom, unnormalized weighted scatter to out
__global__ void __launch_bounds__(256)
edge_scatter_kernel(const float* __restrict__ logits,
                    const unsigned* __restrict__ mkey,
                    const float* __restrict__ z_src,
                    const int* __restrict__ edge_src,
                    const int* __restrict__ edge_dst,
                    float* __restrict__ denom,
                    float* __restrict__ out,
                    int E) {
  const int lane = threadIdx.x & 31;
  const int gwave  = blockIdx.x * (blockDim.x >> 5) + (threadIdx.x >> 5);
  const int nwaves = gridDim.x * (blockDim.x >> 5);
  for (int e = gwave; e < E; e += nwaves) {
    const int src = edge_src[e];
    const int dst = edge_dst[e];
    const float w = expf(logits[e] - funkey(mkey[dst]));
    if (lane == 0) atomicAdd(denom + dst, w);
    const f2 zv = ((const f2*)(z_src + (size_t)src * 64))[lane];
    atomicAdd(out + (size_t)dst * 64 + 2 * lane,     w * zv.x);
    atomicAdd(out + (size_t)dst * 64 + 2 * lane + 1, w * zv.y);
  }
}

// ---------------------------------------------------------------- finalize
__global__ void finalize_kernel(float* __restrict__ out,
                                const float* __restrict__ denom, int n_dst) {
  int stride = gridDim.x * blockDim.x;
  int total = n_dst * 64;
  for (int i = blockIdx.x * blockDim.x + threadIdx.x; i < total; i += stride)
    out[i] /= fmaxf(denom[i >> 6], 1e-9f);
}

extern "C" void kernel_launch(void* const* d_in, const int* in_sizes, int n_in,
                              void* d_out, int out_size, void* d_ws, size_t ws_size,
                              hipStream_t stream) {
  (void)n_in; (void)out_size; (void)ws_size;
  const float* h      = (const float*)d_in[0];
  const float* o      = (const float*)d_in[1];
  const float* tfidf  = (const float*)d_in[2];
  const float* W_fc   = (const float*)d_in[3];
  const float* W_fc1  = (const float*)d_in[4];
  const float* W_feat = (const float*)d_in[5];
  const float* b_feat = (const float*)d_in[6];
  const float* W_attn = (const float*)d_in[7];
  const int* edge_src = (const int*)d_in[8];
  const int* edge_dst = (const int*)d_in[9];
  float* out = (float*)d_out;

  const int IN_DIM = 128, O_DIM = 300;
  const int n_src = in_sizes[0] / IN_DIM;
  const int n_dst = in_sizes[1] / O_DIM;
  const int E     = in_sizes[8];

  // workspace carve: z_src | z_dst | logits | denom | mkey  (~18.2 MB)
  float* z_src  = (float*)d_ws;
  float* z_dst  = z_src + (size_t)n_src * 64;
  float* logits = z_dst + (size_t)n_dst * 64;
  float* denom  = logits + (size_t)E;
  unsigned* mkey = (unsigned*)(denom + n_dst);

  init_kernel<<<512, 256, 0, stream>>>(out, denom, mkey, n_dst);
  proj_kernel<<<(n_src + 3) / 4, 256, 0, stream>>>(h, W_fc,  z_src, n_src, IN_DIM);
  proj_kernel<<<(n_dst + 3) / 4, 256, 0, stream>>>(o, W_fc1, z_dst, n_dst, O_DIM);
  edge_logits_kernel<<<1024, 256, 0, stream>>>(tfidf, W_feat, b_feat, W_attn,
      z_src, z_dst, edge_src, edge_dst, logits, mkey, E);
  edge_scatter_kernel<<<2048, 256, 0, stream>>>(logits, mkey, z_src, edge_src,
      edge_dst, denom, out, E);
  finalize_kernel<<<1024, 256, 0, stream>>>(out, denom, n_dst);
}
